// Linear_28982439313469
// MI455X (gfx1250) — 8-bit matrix operand order repaired and checked by reading, compiled, not run on hardware
//
#include <hip/hip_runtime.h>
#include <hip/hip_bf16.h>

#define K_DIM 4096
#define N_DIM 16384
#define BM 128
#define BN 256

typedef __attribute__((ext_vector_type(8))) int v8i;

__global__ __launch_bounds__(256) void quant_rows(const float* __restrict__ x,
                                                  signed char* __restrict__ qx,
                                                  float* __restrict__ scales) {
    const int row = blockIdx.x;
    const int tid = threadIdx.x;
    const float4* x4 = (const float4*)(x + (size_t)row * K_DIM);

    float4 v[4];
    float amax = 0.0f;
#pragma unroll
    for (int i = 0; i < 4; ++i) {
        v[i] = x4[tid * 4 + i];
        amax = fmaxf(amax, fmaxf(fmaxf(fabsf(v[i].x), fabsf(v[i].y)),
                                 fmaxf(fabsf(v[i].z), fabsf(v[i].w))));
    }

    __shared__ float red[256];
    red[tid] = amax;
    __syncthreads();
#pragma unroll
    for (int s = 128; s > 0; s >>= 1) {
        if (tid < s) red[tid] = fmaxf(red[tid], red[tid + s]);
        __syncthreads();
    }
    const float scale = fmaxf(red[0] * (1.0f / 127.0f), 1e-12f);
    if (tid == 0) scales[row] = scale;
    const float inv = 1.0f / scale;

    int packed[4];
#pragma unroll
    for (int i = 0; i < 4; ++i) {
        float f[4] = {v[i].x, v[i].y, v[i].z, v[i].w};
        int p = 0;
#pragma unroll
        for (int c = 0; c < 4; ++c) {
            int q = (int)fminf(fmaxf(rintf(f[c] * inv), -128.0f), 127.0f);
            p |= (q & 0xFF) << (8 * c);
        }
        packed[i] = p;
    }
    int4 st = {packed[0], packed[1], packed[2], packed[3]};
    ((int4*)(qx + (size_t)row * K_DIM))[tid] = st;
}

__global__ __launch_bounds__(256) void pack_w(const int* __restrict__ w,
                                              signed char* __restrict__ qw,
                                              long long nElems) {
    const long long gid = (long long)blockIdx.x * 256 + threadIdx.x;
    const long long base = gid * 16;
    if (base >= nElems) return;
    const int4* src = (const int4*)(w + base);
    int out[4];
#pragma unroll
    for (int i = 0; i < 4; ++i) {
        int4 t = src[i];
        out[i] = (t.x & 0xFF) | ((t.y & 0xFF) << 8) | ((t.z & 0xFF) << 16) |
                 ((t.w & 0xFF) << 24);
    }
    int4 st = {out[0], out[1], out[2], out[3]};
    ((int4*)qw)[gid] = st;
}

__global__ __launch_bounds__(256) void gemm_i8(const signed char* __restrict__ A,
                                               const signed char* __restrict__ B,
                                               const float* __restrict__ sIn,
                                               const float* __restrict__ sW,
                                               const __hip_bfloat16* __restrict__ bias,
                                               __hip_bfloat16* __restrict__ out) {
    const int lane = threadIdx.x & 31;
    const int wave = threadIdx.x >> 5;
    const int wm = wave & 1;
    const int wn = wave >> 1;
    const int lmod = lane & 15;
    const int lhi = lane >> 4;

    const int m0 = blockIdx.y * BM + wm * 64;
    const int n0 = blockIdx.x * BN + wn * 64;

    v8i acc[4][4] = {};

    const signed char* aptr[4];
    const signed char* bptr[4];
#pragma unroll
    for (int i = 0; i < 4; ++i)
        aptr[i] = A + (size_t)(m0 + i * 16 + lmod) * K_DIM + lhi * 16;
#pragma unroll
    for (int j = 0; j < 4; ++j)
        bptr[j] = B + (size_t)(n0 + j * 16 + lmod) * K_DIM + lhi * 16;

    for (int k = 0; k < K_DIM; k += 64) {
        v8i a[4], b[4];
#pragma unroll
        for (int i = 0; i < 4; ++i) {
            const int2* p = (const int2*)(aptr[i] + k);
            int2 t0 = p[0], t1 = p[1], t2 = p[4], t3 = p[5];
            v8i av;
            av[0] = t0.x; av[1] = t0.y;
            av[2] = t1.x; av[3] = t1.y;
            av[4] = t2.x; av[5] = t2.y;
            av[6] = t3.x; av[7] = t3.y;
            a[i] = av;
        }
#pragma unroll
        for (int j = 0; j < 4; ++j) {
            const int4* q = (const int4*)(bptr[j] + k);
            int4 u0 = q[0], u1 = q[2];
            v8i bv;
            bv[0] = u0.x; bv[1] = u0.y; bv[2] = u0.z; bv[3] = u0.w;
            bv[4] = u1.x; bv[5] = u1.y; bv[6] = u1.z; bv[7] = u1.w;
            b[j] = bv;
        }
#pragma unroll
        for (int i = 0; i < 4; ++i)
#pragma unroll
            for (int j = 0; j < 4; ++j)
                acc[i][j] = __builtin_amdgcn_wmma_i32_16x16x64_iu8(
                     true, a[i],  true, b[j], acc[i][j],
                     false,  false);
    }

#pragma unroll
    for (int i = 0; i < 4; ++i) {
        const int mbase = m0 + i * 16 + lhi * 8;
        float si[8];
#pragma unroll
        for (int r = 0; r < 8; ++r) si[r] = sIn[mbase + r];
#pragma unroll
        for (int j = 0; j < 4; ++j) {
            const int n = n0 + j * 16 + lmod;
            const float sw = sW[n];
            const float bz = __bfloat162float(bias[n]);
#pragma unroll
            for (int r = 0; r < 8; ++r) {
                const int m = mbase + r;
                float val = (float)acc[i][j][r] * si[r] * sw;
                __hip_bfloat16 t = __float2bfloat16(val);
                out[(size_t)m * N_DIM + n] =
                    __float2bfloat16(__bfloat162float(t) + bz);
            }
        }
    }
}

extern "C" void kernel_launch(void* const* d_in, const int* in_sizes, int n_in,
                              void* d_out, int out_size, void* d_ws, size_t ws_size,
                              hipStream_t stream) {
    const float* x = (const float*)d_in[0];
    const int* qw32 = (const int*)d_in[1];
    const float* wscale = (const float*)d_in[2];
    const __hip_bfloat16* bias = (const __hip_bfloat16*)d_in[3];
    __hip_bfloat16* out = (__hip_bfloat16*)d_out;

    const int M = in_sizes[0] / K_DIM;

    char* ws = (char*)d_ws;
    signed char* qx = (signed char*)ws;
    size_t off = (size_t)M * K_DIM;
    float* sIn = (float*)(ws + off);
    off += ((size_t)M * sizeof(float) + 255) & ~(size_t)255;
    signed char* qw = (signed char*)(ws + off);

    quant_rows<<<M, 256, 0, stream>>>(x, qx, sIn);

    const long long wElems = (long long)N_DIM * K_DIM;
    const int packThreads = (int)(wElems / 16);
    pack_w<<<packThreads / 256, 256, 0, stream>>>(qw32, qw, wElems);

    dim3 grid(N_DIM / BN, M / BM);
    gemm_i8<<<grid, 256, 0, stream>>>(qx, qw, sIn, wscale, bias, out);
}
